// AGCRNCellWithMLP_7249904795894
// MI455X (gfx1250) — matrix operand order repaired and checked by reading, not compiled, not run on hardware
//
#include <hip/hip_runtime.h>
#include <hip/hip_bf16.h>

// ---------------- problem constants ----------------
#define N_NODES 4096
#define M_ROWS  2048
#define ISZ     128           // INPUT_SIZE
#define QDIM    32
#define IN_DIM  257           // 2*ISZ + 1
#define KPAD    288           // IN_DIM padded to a multiple of 32
#define GK      (QDIM * KPAD) // 9216 : flattened (d,i) contraction
#define GKB     (GK + QDIM)   // 9248 : + one 32-wide bias chunk (A = q, B = b)

// WMMA fragment types (probe-confirmed signatures)
typedef __bf16 v16bf __attribute__((ext_vector_type(16)));
typedef float  v8f   __attribute__((ext_vector_type(8)));

// ---------------- workspace layout (bytes, 256B aligned) ----------------
#define OFF_CBT  ((size_t)0)
#define SZ_CBT   ((size_t)KPAD * N_NODES * 2)                 // 2,359,296
#define OFF_WTR  (OFF_CBT + SZ_CBT)
#define SZ_WT    ((size_t)ISZ * GKB * 2)                      // 2,367,488
#define OFF_WTU  (OFF_WTR + SZ_WT)
#define OFF_WTC  (OFF_WTU + SZ_WT)
#define OFF_CG   (OFF_WTC + SZ_WT)
#define SZ_CG    ((size_t)M_ROWS * KPAD * 4)                  // 2,359,296
#define OFF_CN   (OFF_CG + SZ_CG)
#define OFF_R    (OFF_CN + SZ_CG)
#define SZ_RU    ((size_t)M_ROWS * ISZ * 4)                   // 1,048,576
#define OFF_U    (OFF_R + SZ_RU)

// ============================================================
// Build cbT[k][n] = bf16( concat(x,h)[n][k] ), zero pad k>=257
// ============================================================
__global__ __launch_bounds__(256)
void k_build_cbT(const float* __restrict__ x, const float* __restrict__ h,
                 __bf16* __restrict__ cbT) {
  int idx = blockIdx.x * blockDim.x + threadIdx.x;  // over KPAD * N_NODES
  if (idx >= KPAD * N_NODES) return;
  int k = idx / N_NODES;
  int n = idx - k * N_NODES;
  float v = 0.0f;
  if (k < ISZ + 1)       v = x[(size_t)n * (ISZ + 1) + k];
  else if (k < IN_DIM)   v = h[(size_t)n * ISZ + (k - (ISZ + 1))];
  cbT[(size_t)k * N_NODES + n] = (__bf16)v;
}

// ============================================================
// Build Wt[o][kk] : kk < GK -> W[d][i][o] (i<257, else 0),
//                   kk >= GK -> b[kk-GK][o]  (bias chunk)
// ============================================================
__global__ __launch_bounds__(256)
void k_build_Wt(const float* __restrict__ W, const float* __restrict__ b,
                __bf16* __restrict__ Wt) {
  int idx = blockIdx.x * blockDim.x + threadIdx.x;  // over ISZ * GKB
  if (idx >= ISZ * GKB) return;
  int o  = idx / GKB;
  int kk = idx - o * GKB;
  float v = 0.0f;
  if (kk < GK) {
    int d = kk / KPAD;
    int i = kk - d * KPAD;
    if (i < IN_DIM) v = W[((size_t)d * IN_DIM + i) * ISZ + o];
  } else {
    v = b[(size_t)(kk - GK) * ISZ + o];
  }
  Wt[(size_t)o * GKB + kk] = (__bf16)v;
}

// ============================================================
// GEMM1: cg[m][kcol] = sum_n adj[nodes[m]][n] * cb[n][kcol]
// One wave computes a 16(M) x 48(N) block: 3 N-tiles sharing one A fragment
// per K chunk. K = 4096 over n, bf16 WMMA, f32 acc.
// A-frag layout (16-bit A, ISA 7.12.2): lane = half*16 + r holds row M=r,
//   element e: K = half*16 + e   -> contiguous, same order as B (not the ISA table order).
// B-frag: lane = half*16 + c holds col N=c, element e: K = half*16 + e.
// ============================================================
#define ADJ_NT 3   // N tiles per wave; 288/16/3 = 6 wave-groups in N
__global__ __launch_bounds__(256)
void k_gemm_adj(const float* __restrict__ adj, const int* __restrict__ nodes,
                const __bf16* __restrict__ cbT, float* __restrict__ cg) {
  const int wave  = (blockIdx.x * blockDim.x + threadIdx.x) >> 5;
  const int lane  = threadIdx.x & 31;
  const int half  = lane >> 4;
  const int lr    = lane & 15;
  const int tileM = wave & 127;          // 2048/16 = 128 tiles in M
  const int ngrp  = wave >> 7;           // 0..5, each covers ADJ_NT tiles

  const int m    = tileM * 16 + lr;
  const int rowm = nodes[m];
  const float*  __restrict__ arow  = adj + (size_t)rowm * N_NODES;
  // lane's B column for tile t is (ngrp*ADJ_NT + t)*16 + lr
  const __bf16* __restrict__ bbase =
      cbT + (size_t)(ngrp * (ADJ_NT * 16) + lr) * N_NODES;

  v8f acc[ADJ_NT];
#pragma unroll
  for (int t = 0; t < ADJ_NT; ++t) acc[t] = (v8f){};

  for (int k0 = 0; k0 < N_NODES; k0 += 32) {
    // shared A fragment (fp32 -> bf16 on the fly)
    v16bf a;
    const float* p0 = arow + k0 + half * 16;      // K = half*16 + e
    const float* p1 = p0 + 8;                     // K = half*16 + 8 + e
#pragma unroll
    for (int e = 0; e < 8; ++e) {
      a[e]     = (__bf16)p0[e];
      a[8 + e] = (__bf16)p1[e];
    }
    __builtin_prefetch(arow + k0 + 128, 0, 0);    // stream next A chunks
#pragma unroll
    for (int t = 0; t < ADJ_NT; ++t) {
      v16bf bf = *(const v16bf*)(bbase + (size_t)t * 16 * N_NODES +
                                 k0 + half * 16);
      acc[t] = __builtin_amdgcn_wmma_f32_16x16x32_bf16(false, a, false, bf,
                                                       (short)0, acc[t],
                                                       false, false);
    }
  }
#pragma unroll
  for (int t = 0; t < ADJ_NT; ++t) {
    const int col = (ngrp * ADJ_NT + t) * 16 + lr;
#pragma unroll
    for (int j = 0; j < 8; ++j) {
      int mo = tileM * 16 + half * 8 + j;         // C/D layout: M = half*8 + j
      cg[(size_t)mo * KPAD + col] = acc[t][j];
    }
  }
}

// ============================================================
// Gate GEMM: dst[m][o] = act( sum_{d,i} q[m,d]*src[m,i]*W[d,i,o] + (q@b)[m,o] )
// One wave computes a 16(M) x 64(O) block: 4 O-tiles sharing one generated
// A fragment per K chunk (A-gen is the expensive part: qv*src -> bf16).
// chunk kc<288: A = q[m, kc/9] * src[m, (kc%9)*32 + ..]
// chunk kc==288 (bias): A = q row, B rows = b.
// mode 0 = sigmoid (r,u gates), mode 1 = tanh (candidate)
// ============================================================
#define GATE_NT 4  // O tiles per wave; 128/16/4 = 2 wave-groups in O
__global__ __launch_bounds__(256)
void k_gemm_gate(const float* __restrict__ src, const float* __restrict__ q,
                 const __bf16* __restrict__ Wt, float* __restrict__ dst, int mode) {
  const int wave  = (blockIdx.x * blockDim.x + threadIdx.x) >> 5;
  const int lane  = threadIdx.x & 31;
  const int half  = lane >> 4;
  const int lr    = lane & 15;
  const int ogrp  = wave & 1;            // 0..1, each covers GATE_NT O-tiles
  const int tileM = wave >> 1;           // 128 tiles in M

  const int m = tileM * 16 + lr;
  const float* __restrict__ srow = src + (size_t)m * KPAD;
  const float* __restrict__ qrow = q   + (size_t)m * QDIM;
  // lane's B column for tile t is ogrp*64 + t*16 + lr
  const __bf16* __restrict__ wbase = Wt + (size_t)(ogrp * 64 + lr) * GKB;

  v8f acc[GATE_NT];
#pragma unroll
  for (int t = 0; t < GATE_NT; ++t) acc[t] = (v8f){};

  for (int kc = 0; kc < GKB / 32; ++kc) {          // 289 chunks of K=32
    v16bf a;
    if (kc < GK / 32) {
      const int d  = kc / (KPAD / 32);             // which q element
      const int i0 = (kc - d * (KPAD / 32)) * 32;  // base i within src row
      const float qv = qrow[d];
      const float* p0 = srow + i0 + half * 16;
      const float* p1 = p0 + 8;
#pragma unroll
      for (int e = 0; e < 8; ++e) {
        a[e]     = (__bf16)(qv * p0[e]);
        a[8 + e] = (__bf16)(qv * p1[e]);
      }
    } else {                                       // bias chunk: A = q row
      const float* p0 = qrow + half * 16;
      const float* p1 = p0 + 8;
#pragma unroll
      for (int e = 0; e < 8; ++e) {
        a[e]     = (__bf16)p0[e];
        a[8 + e] = (__bf16)p1[e];
      }
    }
#pragma unroll
    for (int t = 0; t < GATE_NT; ++t) {
      v16bf bf = *(const v16bf*)(wbase + (size_t)t * 16 * GKB +
                                 kc * 32 + half * 16);
      acc[t] = __builtin_amdgcn_wmma_f32_16x16x32_bf16(false, a, false, bf,
                                                       (short)0, acc[t],
                                                       false, false);
    }
  }
#pragma unroll
  for (int t = 0; t < GATE_NT; ++t) {
    const int o = ogrp * 64 + t * 16 + lr;
#pragma unroll
    for (int j = 0; j < 8; ++j) {
      int mo = tileM * 16 + half * 8 + j;
      float v = acc[t][j];
      v = (mode == 0) ? (1.0f / (1.0f + __expf(-v))) : tanhf(v);
      dst[(size_t)mo * ISZ + o] = v;
    }
  }
}

// ============================================================
// cn[m][k] = concat(x, r*h)[nodes[m]][k]  (raw x/h, NOT adj-convolved)
// ============================================================
__global__ __launch_bounds__(256)
void k_build_cn(const float* __restrict__ x, const float* __restrict__ h,
                const int* __restrict__ nodes, const float* __restrict__ r,
                float* __restrict__ cn) {
  int idx = blockIdx.x * blockDim.x + threadIdx.x;   // over M_ROWS * KPAD
  if (idx >= M_ROWS * KPAD) return;
  int mrow = idx / KPAD;
  int k    = idx - mrow * KPAD;
  int nid  = nodes[mrow];
  float v = 0.0f;
  if (k < ISZ + 1)      v = x[(size_t)nid * (ISZ + 1) + k];
  else if (k < IN_DIM) {
    int j = k - (ISZ + 1);
    v = r[(size_t)mrow * ISZ + j] * h[(size_t)nid * ISZ + j];
  }
  cn[idx] = v;
}

// ============================================================
// new_h = (1-u) * (r*h[nodes]) + u * cand     (cand already in d_out)
// ============================================================
__global__ __launch_bounds__(256)
void k_final(const float* __restrict__ h, const int* __restrict__ nodes,
             const float* __restrict__ r, const float* __restrict__ u,
             float* __restrict__ out) {
  int idx = blockIdx.x * blockDim.x + threadIdx.x;   // over M_ROWS * ISZ
  if (idx >= M_ROWS * ISZ) return;
  int mrow = idx / ISZ;
  int j    = idx - mrow * ISZ;
  float hv = h[(size_t)nodes[mrow] * ISZ + j];
  float rv = r[idx];
  float uv = u[idx];
  float cd = out[idx];
  out[idx] = (1.0f - uv) * rv * hv + uv * cd;
}

// ============================================================
extern "C" void kernel_launch(void* const* d_in, const int* in_sizes, int n_in,
                              void* d_out, int out_size, void* d_ws, size_t ws_size,
                              hipStream_t stream) {
  const float* x     = (const float*)d_in[0];
  const float* h     = (const float*)d_in[1];
  const float* q     = (const float*)d_in[2];
  const float* adj   = (const float*)d_in[3];
  const int*   nodes = (const int*)  d_in[4];
  const float* W_u   = (const float*)d_in[5];
  const float* b_u   = (const float*)d_in[6];
  const float* W_r   = (const float*)d_in[7];
  const float* b_r   = (const float*)d_in[8];
  const float* W_c   = (const float*)d_in[9];
  const float* b_c   = (const float*)d_in[10];
  float* out = (float*)d_out;

  char* ws = (char*)d_ws;
  __bf16* cbT  = (__bf16*)(ws + OFF_CBT);
  __bf16* Wt_r = (__bf16*)(ws + OFF_WTR);
  __bf16* Wt_u = (__bf16*)(ws + OFF_WTU);
  __bf16* Wt_c = (__bf16*)(ws + OFF_WTC);
  float*  cg   = (float*) (ws + OFF_CG);
  float*  cn   = (float*) (ws + OFF_CN);
  float*  rbuf = (float*) (ws + OFF_R);
  float*  ubuf = (float*) (ws + OFF_U);

  // 1) operand prep (fp32 -> padded bf16, K-contiguous B layouts)
  k_build_cbT<<<(KPAD * N_NODES + 255) / 256, 256, 0, stream>>>(x, h, cbT);
  k_build_Wt<<<(ISZ * GKB + 255) / 256, 256, 0, stream>>>(W_r, b_r, Wt_r);
  k_build_Wt<<<(ISZ * GKB + 255) / 256, 256, 0, stream>>>(W_u, b_u, Wt_u);
  k_build_Wt<<<(ISZ * GKB + 255) / 256, 256, 0, stream>>>(W_c, b_c, Wt_c);

  // 2) graph convolution: cg = adj[nodes] @ concat(x,h)
  //    128 M-tiles * 6 N-groups = 768 waves -> 96 blocks of 8 waves
  k_gemm_adj<<<(128 * 6) / 8, 256, 0, stream>>>(adj, nodes, cbT, cg);

  // 3) r and u gates (sigmoid): 128 M-tiles * 2 O-groups = 256 waves -> 32 blocks
  k_gemm_gate<<<(128 * 2) / 8, 256, 0, stream>>>(cg, q, Wt_r, rbuf, 0);
  k_gemm_gate<<<(128 * 2) / 8, 256, 0, stream>>>(cg, q, Wt_u, ubuf, 0);

  // 4) candidate input from raw x and r*h
  k_build_cn<<<(M_ROWS * KPAD + 255) / 256, 256, 0, stream>>>(x, h, nodes, rbuf, cn);

  // 5) candidate gate (tanh) straight into d_out
  k_gemm_gate<<<(128 * 2) / 8, 256, 0, stream>>>(cn, q, Wt_c, out, 1);

  // 6) blend: out = (1-u)*r*h + u*cand
  k_final<<<(M_ROWS * ISZ + 255) / 256, 256, 0, stream>>>(h, nodes, rbuf, ubuf, out);
}
